// DecoderCUGOSum_42777874268719
// MI455X (gfx1250) — matrix operand order repaired and checked by reading, not compiled, not run on hardware
//
#include <hip/hip_runtime.h>
#include <hip/hip_bf16.h>

// MI455X / gfx1250 fused GNN decoder (m2g).  All GEMMs via v_wmma_f32_16x16x32_f16,
// f32 accumulate; activations / LayerNorm / segment-sum fused into GEMM epilogues.
// Round 3: B-fragments batched per k-chunk into explicit arrays so the 16
// ds_load_b128 clause-drain ahead of the 8-WMMA burst (no per-WMMA dscnt-0 stall).

#define DD 128
#define HH 128

typedef __attribute__((ext_vector_type(16))) _Float16 v16h;
typedef __attribute__((ext_vector_type(8)))  _Float16 v8h;
typedef __attribute__((ext_vector_type(8)))  float    v8f;

__device__ __forceinline__ v8f wmma_f16(v16h a, v16h b, v8f c) {
  return __builtin_amdgcn_wmma_f32_16x16x32_f16(
      /*neg_a=*/false, a, /*neg_b=*/false, b,
      /*c_mod=*/(short)0, c, /*reuse_a=*/false, /*reuse_b=*/false);
}

// A fragment (16x32, f16) from an f32 row-major row pointer.
// ISA layout: lane<16 holds K = {k0..k0+7, k0+16..k0+23}; lane>=16 shifts +8.
__device__ __forceinline__ v16h a_frag_f32(const float* __restrict__ row, int k0, int hi) {
  const float4* p0 = (const float4*)(row + k0 + hi * 8);
  const float4* p1 = (const float4*)(row + k0 + 16 + hi * 8);
  float4 x0 = p0[0], x1 = p0[1];
  float4 y0 = p1[0], y1 = p1[1];
  v16h a;
  a[0]  = (_Float16)x0.x; a[1]  = (_Float16)x0.y; a[2]  = (_Float16)x0.z; a[3]  = (_Float16)x0.w;
  a[4]  = (_Float16)x1.x; a[5]  = (_Float16)x1.y; a[6]  = (_Float16)x1.z; a[7]  = (_Float16)x1.w;
  a[8]  = (_Float16)y0.x; a[9]  = (_Float16)y0.y; a[10] = (_Float16)y0.z; a[11] = (_Float16)y0.w;
  a[12] = (_Float16)y1.x; a[13] = (_Float16)y1.y; a[14] = (_Float16)y1.z; a[15] = (_Float16)y1.w;
  return a;
}

// A fragment from an f16 row (LDS h-stage), same K layout.
__device__ __forceinline__ v16h a_frag_h16(const _Float16* __restrict__ row, int k0, int hi) {
  v8h lo = *(const v8h*)(row + k0 + hi * 8);
  v8h hv = *(const v8h*)(row + k0 + 16 + hi * 8);
  v16h a;
#pragma unroll
  for (int j = 0; j < 8; ++j) { a[j] = lo[j]; a[8 + j] = hv[j]; }
  return a;
}

// B fragment (32x16, f16) from transposed weight wt[n][k] (leading dim ldk), in LDS.
// Lanes 0-15: N=lane, K={k0..k0+7, k0+16..k0+23}; lanes 16-31: N=lane-16, K shifted +8.
__device__ __forceinline__ v16h b_frag(const _Float16* __restrict__ wt, int ldk,
                                       int n0, int k0, int lane) {
  int n = n0 + (lane & 15);
  int k = k0 + ((lane >> 4) << 3);
  const _Float16* p = wt + (size_t)n * ldk + k;
  v8h lo = *(const v8h*)p;
  v8h hv = *(const v8h*)(p + 16);
  v16h b;
#pragma unroll
  for (int j = 0; j < 8; ++j) { b[j] = lo[j]; b[8 + j] = hv[j]; }
  return b;
}

// Sum across the 16 lanes of a half-wave (C-layout rows live in lane groups of 16).
__device__ __forceinline__ float hsum16(float v) {
  v += __shfl_xor(v, 1, 32);
  v += __shfl_xor(v, 2, 32);
  v += __shfl_xor(v, 4, 32);
  v += __shfl_xor(v, 8, 32);
  return v;
}

__device__ __forceinline__ float silu_f(float x) { return x / (1.f + __expf(-x)); }

// ---------------- weight transpose + f16 convert ----------------
__global__ __launch_bounds__(256) void prep_weights(
    const float* __restrict__ We, const float* __restrict__ Ws, const float* __restrict__ Wd,
    const float* __restrict__ W1e, const float* __restrict__ Wn0, const float* __restrict__ Wn1,
    _Float16* __restrict__ wet, _Float16* __restrict__ wst, _Float16* __restrict__ wdt,
    _Float16* __restrict__ w1et, _Float16* __restrict__ wn0t, _Float16* __restrict__ wn1t) {
  int i = blockIdx.x * blockDim.x + threadIdx.x;
  if (i < DD * HH) {
    int k = i >> 7, n = i & 127;
    int o = n * DD + k;
    wet[o]  = (_Float16)We[i];
    wst[o]  = (_Float16)Ws[i];
    wdt[o]  = (_Float16)Wd[i];
    w1et[o] = (_Float16)W1e[i];
    wn1t[o] = (_Float16)Wn1[i];
  }
  if (i < 2 * DD * HH) {
    int k = i >> 7, n = i & 127;
    wn0t[n * (2 * DD) + k] = (_Float16)Wn0[i];
  }
}

__global__ __launch_bounds__(256) void zero_f32(float4* __restrict__ p, size_t n4) {
  size_t i = (size_t)blockIdx.x * blockDim.x + threadIdx.x;
  if (i < n4) p[i] = make_float4(0.f, 0.f, 0.f, 0.f);
}

// ---------------- projection GEMM: Y[f16] = X[f32] @ W (K=128) ----------------
__global__ __launch_bounds__(256) void proj_kernel(
    const float* __restrict__ X, const _Float16* __restrict__ wt_g,
    _Float16* __restrict__ Y, int nrows) {
  __shared__ __align__(32) _Float16 wt[DD * HH];           // 32 KB
  int tid = threadIdx.x, lane = tid & 31, wave = tid >> 5;
  for (int i = tid * 8; i < DD * HH; i += 256 * 8)
    *(v8h*)(wt + i) = *(const v8h*)(wt_g + i);
  __syncthreads();
  int hi = lane >> 4, nl = lane & 15;

  for (int tb = blockIdx.x; tb * 128 < nrows; tb += gridDim.x) {
    int row0 = tb * 128 + wave * 16;
    if (row0 >= nrows) continue;                           // wave-uniform

    // prefetch next tile's A row for this lane
    int rn0 = (tb + gridDim.x) * 128 + wave * 16;
    if (rn0 < nrows) {
      const float* pn = X + (size_t)(rn0 + nl) * DD;
      __builtin_prefetch(pn, 0, 0);       __builtin_prefetch(pn + 32, 0, 0);
      __builtin_prefetch(pn + 64, 0, 0);  __builtin_prefetch(pn + 96, 0, 0);
    }

    int r = row0 + nl;
    int rc = r < nrows - 1 ? r : nrows - 1;                // clamp (mesh rows not /16)
    const float* arow = X + (size_t)rc * DD;
    v16h af[4];
#pragma unroll
    for (int kc = 0; kc < 4; ++kc) af[kc] = a_frag_f32(arow, kc * 32, hi);
    v8f acc[8] = {};
#pragma unroll
    for (int kc = 0; kc < 4; ++kc) {
      v16h bf[8];
#pragma unroll
      for (int t = 0; t < 8; ++t) bf[t] = b_frag(wt, DD, t * 16, kc * 32, lane);
#pragma unroll
      for (int t = 0; t < 8; ++t) acc[t] = wmma_f16(af[kc], bf[t], acc[t]);
    }

#pragma unroll
    for (int i = 0; i < 8; ++i) {
      int m = row0 + i + 8 * hi;
      if (m < nrows) {
        _Float16* yrow = Y + (size_t)m * HH;
#pragma unroll
        for (int t = 0; t < 8; ++t) yrow[t * 16 + nl] = (_Float16)acc[t][i];
      }
    }
  }
}

// ---------------- fused edge MLP + LayerNorm + segment-sum scatter ----------------
__global__ __launch_bounds__(256) void edge_kernel(
    const float* __restrict__ efeat, const int* __restrict__ src_idx,
    const int* __restrict__ dst_idx,
    const _Float16* __restrict__ wet_g, const _Float16* __restrict__ w1et_g,
    const _Float16* __restrict__ ps, const _Float16* __restrict__ pd,
    const float* __restrict__ b0, const float* __restrict__ b1e,
    const float* __restrict__ ge, const float* __restrict__ be,
    float* __restrict__ agg, int nedges) {
  extern __shared__ __align__(32) _Float16 smem_e[];       // 96 KB dynamic
  _Float16* wet    = smem_e;                               // [128][128]
  _Float16* w1t    = smem_e + DD * HH;                     // [128][128]
  _Float16* hstage = smem_e + 2 * DD * HH;                 // 8 waves * 16 * 128
  int tid = threadIdx.x, lane = tid & 31, wave = tid >> 5;
  for (int i = tid * 8; i < DD * HH; i += 256 * 8) {
    *(v8h*)(wet + i) = *(const v8h*)(wet_g + i);
    *(v8h*)(w1t + i) = *(const v8h*)(w1et_g + i);
  }
  __syncthreads();
  int hi = lane >> 4, nl = lane & 15;

  float b0v[8], b1v[8], gev[8], bev[8];
#pragma unroll
  for (int t = 0; t < 8; ++t) {
    int n = t * 16 + nl;
    b0v[t] = b0[n]; b1v[t] = b1e[n]; gev[t] = ge[n]; bev[t] = be[n];
  }
  _Float16* hbase = hstage + (size_t)wave * 16 * DD;
  const _Float16* harow = hbase + (size_t)nl * DD;

  for (int tb = blockIdx.x; tb * 128 < nedges; tb += gridDim.x) {
    int e0 = tb * 128 + wave * 16;
    if (e0 >= nedges) continue;                            // wave-uniform (E % 16 == 0)

    int en0 = (tb + gridDim.x) * 128 + wave * 16;
    if (en0 < nedges) {
      const float* pn = efeat + (size_t)(en0 + nl) * DD;
      __builtin_prefetch(pn, 0, 0);       __builtin_prefetch(pn + 32, 0, 0);
      __builtin_prefetch(pn + 64, 0, 0);  __builtin_prefetch(pn + 96, 0, 0);
    }

    // GEMM1: X = efeat @ We
    const float* arow = efeat + (size_t)(e0 + nl) * DD;
    v16h af[4];
#pragma unroll
    for (int kc = 0; kc < 4; ++kc) af[kc] = a_frag_f32(arow, kc * 32, hi);
    v8f acc[8] = {};
#pragma unroll
    for (int kc = 0; kc < 4; ++kc) {
      v16h bf[8];
#pragma unroll
      for (int t = 0; t < 8; ++t) bf[t] = b_frag(wet, DD, t * 16, kc * 32, lane);
#pragma unroll
      for (int t = 0; t < 8; ++t) acc[t] = wmma_f16(af[kc], bf[t], acc[t]);
    }

    // gather-add + SiLU, stage h into LDS (transpose C-layout -> row-major f16)
#pragma unroll
    for (int i = 0; i < 8; ++i) {
      int e = e0 + i + 8 * hi;
      const _Float16* psr = ps + (size_t)src_idx[e] * HH;
      const _Float16* pdr = pd + (size_t)dst_idx[e] * HH;
      _Float16* hrow = hbase + (size_t)(i + 8 * hi) * DD;
#pragma unroll
      for (int t = 0; t < 8; ++t) {
        int n = t * 16 + nl;
        float x = acc[t][i] + (float)psr[n] + (float)pdr[n] + b0v[t];
        hrow[n] = (_Float16)silu_f(x);
      }
    }

    // GEMM2: Y = h @ W1e  (A re-read from LDS in A-fragment layout)
    v8f acc2[8] = {};
#pragma unroll
    for (int kc = 0; kc < 4; ++kc) {
      v16h a = a_frag_h16(harow, kc * 32, hi);
      v16h bf[8];
#pragma unroll
      for (int t = 0; t < 8; ++t) bf[t] = b_frag(w1t, DD, t * 16, kc * 32, lane);
#pragma unroll
      for (int t = 0; t < 8; ++t) acc2[t] = wmma_f16(a, bf[t], acc2[t]);
    }

    // LayerNorm (rows live in half-wave lane groups) + atomic segment-sum
#pragma unroll
    for (int i = 0; i < 8; ++i) {
      float s = 0.f, q = 0.f;
#pragma unroll
      for (int t = 0; t < 8; ++t) {
        float y = acc2[t][i] + b1v[t];
        acc2[t][i] = y;
        s += y; q += y * y;
      }
      s = hsum16(s); q = hsum16(q);
      float mu = s * (1.f / 128.f);
      float rstd = rsqrtf(q * (1.f / 128.f) - mu * mu + 1e-5f);
      int e = e0 + i + 8 * hi;
      float* arow_out = agg + (size_t)dst_idx[e] * DD;
#pragma unroll
      for (int t = 0; t < 8; ++t) {
        float y = (acc2[t][i] - mu) * rstd * gev[t] + bev[t];
        atomicAdd(arow_out + t * 16 + nl, y);
      }
    }
  }
}

// ---------------- fused node MLP (K=256) + LayerNorm + residual ----------------
__global__ __launch_bounds__(256) void node_kernel(
    const float* __restrict__ grid, const float* __restrict__ aggv,
    const _Float16* __restrict__ wn0t_g, const _Float16* __restrict__ wn1t_g,
    const float* __restrict__ bn0, const float* __restrict__ bn1,
    const float* __restrict__ gn, const float* __restrict__ bnv_,
    float* __restrict__ out, int nrows) {
  extern __shared__ __align__(32) _Float16 smem_n[];       // 128 KB dynamic
  _Float16* w0t    = smem_n;                               // [128][256]
  _Float16* w1t    = smem_n + 2 * DD * HH;                 // [128][128]
  _Float16* hstage = smem_n + 3 * DD * HH;
  int tid = threadIdx.x, lane = tid & 31, wave = tid >> 5;
  for (int i = tid * 8; i < 2 * DD * HH; i += 256 * 8)
    *(v8h*)(w0t + i) = *(const v8h*)(wn0t_g + i);
  for (int i = tid * 8; i < DD * HH; i += 256 * 8)
    *(v8h*)(w1t + i) = *(const v8h*)(wn1t_g + i);
  __syncthreads();
  int hi = lane >> 4, nl = lane & 15;

  float b0v[8], b1v[8], gv[8], bv[8];
#pragma unroll
  for (int t = 0; t < 8; ++t) {
    int n = t * 16 + nl;
    b0v[t] = bn0[n]; b1v[t] = bn1[n]; gv[t] = gn[n]; bv[t] = bnv_[n];
  }
  _Float16* hbase = hstage + (size_t)wave * 16 * DD;
  const _Float16* harow = hbase + (size_t)nl * DD;

  for (int tb = blockIdx.x; tb * 128 < nrows; tb += gridDim.x) {
    int r0 = tb * 128 + wave * 16;
    if (r0 >= nrows) continue;                             // wave-uniform

    int rn0 = (tb + gridDim.x) * 128 + wave * 16;
    if (rn0 < nrows) {
      const float* pn = grid + (size_t)(rn0 + nl) * DD;
      const float* qn = aggv + (size_t)(rn0 + nl) * DD;
      __builtin_prefetch(pn, 0, 0); __builtin_prefetch(pn + 64, 0, 0);
      __builtin_prefetch(qn, 0, 0); __builtin_prefetch(qn + 64, 0, 0);
    }

    // GEMM1: hn_pre = [grid, agg] @ Wn0   (K = 256, two source matrices)
    const float* grow = grid + (size_t)(r0 + nl) * DD;
    const float* arow = aggv + (size_t)(r0 + nl) * DD;
    v8f acc[8] = {};
#pragma unroll
    for (int kc = 0; kc < 4; ++kc) {
      v16h a = a_frag_f32(grow, kc * 32, hi);
      v16h bf[8];
#pragma unroll
      for (int t = 0; t < 8; ++t) bf[t] = b_frag(w0t, 2 * DD, t * 16, kc * 32, lane);
#pragma unroll
      for (int t = 0; t < 8; ++t) acc[t] = wmma_f16(a, bf[t], acc[t]);
    }
#pragma unroll
    for (int kc = 0; kc < 4; ++kc) {
      v16h a = a_frag_f32(arow, kc * 32, hi);
      v16h bf[8];
#pragma unroll
      for (int t = 0; t < 8; ++t) bf[t] = b_frag(w0t, 2 * DD, t * 16, DD + kc * 32, lane);
#pragma unroll
      for (int t = 0; t < 8; ++t) acc[t] = wmma_f16(a, bf[t], acc[t]);
    }

    // SiLU + stage
#pragma unroll
    for (int i = 0; i < 8; ++i) {
      _Float16* hrow = hbase + (size_t)(i + 8 * hi) * DD;
#pragma unroll
      for (int t = 0; t < 8; ++t) {
        float x = acc[t][i] + b0v[t];
        hrow[t * 16 + nl] = (_Float16)silu_f(x);
      }
    }

    // GEMM2: hn @ Wn1
    v8f acc2[8] = {};
#pragma unroll
    for (int kc = 0; kc < 4; ++kc) {
      v16h a = a_frag_h16(harow, kc * 32, hi);
      v16h bf[8];
#pragma unroll
      for (int t = 0; t < 8; ++t) bf[t] = b_frag(w1t, DD, t * 16, kc * 32, lane);
#pragma unroll
      for (int t = 0; t < 8; ++t) acc2[t] = wmma_f16(a, bf[t], acc2[t]);
    }

    // LayerNorm + residual, store
#pragma unroll
    for (int i = 0; i < 8; ++i) {
      float s = 0.f, q = 0.f;
#pragma unroll
      for (int t = 0; t < 8; ++t) {
        float y = acc2[t][i] + b1v[t];
        acc2[t][i] = y;
        s += y; q += y * y;
      }
      s = hsum16(s); q = hsum16(q);
      float mu = s * (1.f / 128.f);
      float rstd = rsqrtf(q * (1.f / 128.f) - mu * mu + 1e-5f);
      int m = r0 + i + 8 * hi;
      const float* gr = grid + (size_t)m * DD;
      float* orow = out + (size_t)m * DD;
#pragma unroll
      for (int t = 0; t < 8; ++t) {
        int n = t * 16 + nl;
        orow[n] = (acc2[t][i] - mu) * rstd * gv[t] + bv[t] + gr[n];
      }
    }
  }
}

extern "C" void kernel_launch(void* const* d_in, const int* in_sizes, int n_in,
                              void* d_out, int out_size, void* d_ws, size_t ws_size,
                              hipStream_t stream) {
  const int N_MESH = 40962, N_GRID = 200000, N_EDGES = 600000;

  const float* m2g_efeat = (const float*)d_in[0];
  const float* grid_nfeat = (const float*)d_in[1];
  const float* mesh_nfeat = (const float*)d_in[2];
  const int*   src_idx    = (const int*)d_in[3];
  const int*   dst_idx    = (const int*)d_in[4];
  const float* We  = (const float*)d_in[5];
  const float* Ws  = (const float*)d_in[6];
  const float* Wd  = (const float*)d_in[7];
  const float* b0  = (const float*)d_in[8];
  const float* W1e = (const float*)d_in[9];
  const float* b1e = (const float*)d_in[10];
  const float* ge  = (const float*)d_in[11];
  const float* be  = (const float*)d_in[12];
  const float* Wn0 = (const float*)d_in[13];
  const float* bn0 = (const float*)d_in[14];
  const float* Wn1 = (const float*)d_in[15];
  const float* bn1 = (const float*)d_in[16];
  const float* gn  = (const float*)d_in[17];
  const float* bn  = (const float*)d_in[18];

  // workspace layout (all offsets 32B-aligned)
  _Float16* ps  = (_Float16*)d_ws;                        // [N_MESH,128] f16
  _Float16* pd  = ps + (size_t)N_MESH * 128;              // [N_GRID,128] f16
  float*    agg = (float*)(pd + (size_t)N_GRID * 128);    // [N_GRID,128] f32
  _Float16* wet  = (_Float16*)(agg + (size_t)N_GRID * 128);
  _Float16* wst  = wet  + 128 * 128;
  _Float16* wdt  = wst  + 128 * 128;
  _Float16* w1et = wdt  + 128 * 128;
  _Float16* wn1t = w1et + 128 * 128;
  _Float16* wn0t = wn1t + 128 * 128;                      // [128][256]

  prep_weights<<<(2 * 128 * 128 + 255) / 256, 256, 0, stream>>>(
      We, Ws, Wd, W1e, Wn0, Wn1, wet, wst, wdt, w1et, wn0t, wn1t);

  size_t agg4 = (size_t)N_GRID * 128 / 4;
  zero_f32<<<(int)((agg4 + 255) / 256), 256, 0, stream>>>((float4*)agg, agg4);

  int mesh_tiles = (N_MESH + 127) / 128;                  // 321
  int grid_tiles = (N_GRID + 127) / 128;                  // 1563
  int edge_tiles = (N_EDGES + 127) / 128;                 // 4688
  int grid_blocks = (grid_tiles + 3) / 4;                 // 391
  int edge_blocks = (edge_tiles + 3) / 4;                 // 1172

  proj_kernel<<<mesh_tiles, 256, 0, stream>>>(mesh_nfeat, wst, ps, N_MESH);
  proj_kernel<<<grid_blocks, 256, 0, stream>>>(grid_nfeat, wdt, pd, N_GRID);

  edge_kernel<<<edge_blocks, 256, 3 * DD * HH * (int)sizeof(_Float16), stream>>>(
      m2g_efeat, src_idx, dst_idx, wet, w1et, ps, pd, b0, b1e, ge, be, agg, N_EDGES);

  node_kernel<<<grid_blocks, 256, 4 * DD * HH * (int)sizeof(_Float16), stream>>>(
      grid_nfeat, agg, wn0t, wn1t, bn0, bn1, gn, bn, (float*)d_out, N_GRID);
}
